// MultiHeadAttention_11338713661451
// MI455X (gfx1250) — matrix operand order repaired and checked by reading, not compiled, not run on hardware
//
#include <hip/hip_runtime.h>

// ---------------------------------------------------------------------------
// MultiHeadAttention for MI455X (gfx1250, wave32, WMMA)
// B=2, L=2048, E=1024, H=16, D=64
//   1) fp32 -> bf16 convert (X, Wq, Wk, Wv, Wo)
//   2) Q = X Wq^T, K = X Wk^T  -> bf16 [B,H,L,D]
//      V = X Wv^T              -> bf16 [B,H,D,L] (pre-transposed for PV NT-GEMM)
//   3) flash attention per (b,h), bf16 WMMA w/ fp32 accum -> bf16 [B,L,E]
//   4) out = Attn Wo^T + bo    -> fp32 d_out
// Global->LDS traffic uses GLOBAL_LOAD_ASYNC_TO_LDS_B128 (ASYNCcnt) when the
// toolchain declares the builtin, with a sync load+ds_store fallback.
// ---------------------------------------------------------------------------

typedef __attribute__((ext_vector_type(16))) __bf16 bf16x16;
typedef __attribute__((ext_vector_type(8)))  float  floatx8;
typedef __attribute__((ext_vector_type(4)))  int    intx4;

#ifndef __has_builtin
#define __has_builtin(x) 0
#endif
#if __has_builtin(__builtin_amdgcn_global_load_async_to_lds_b128)
#define HAVE_ASYNC_LDS 1
#else
#define HAVE_ASYNC_LDS 0
#endif

// Copy 32 contiguous bytes global -> LDS (two B128 transfers).
static __device__ __forceinline__ void g2l_32B(__bf16* l, const __bf16* g) {
#if HAVE_ASYNC_LDS
    typedef __attribute__((address_space(1))) intx4 gv4i;   // global int4
    typedef __attribute__((address_space(3))) intx4 lv4i;   // LDS int4
    gv4i* gp0 = (gv4i*)(uintptr_t)(g);
    gv4i* gp1 = (gv4i*)(uintptr_t)(g + 8);
    lv4i* lp0 = (lv4i*)(unsigned)(uintptr_t)(l);
    lv4i* lp1 = (lv4i*)(unsigned)(uintptr_t)(l + 8);
    __builtin_amdgcn_global_load_async_to_lds_b128(gp0, lp0, 0, 0);
    __builtin_amdgcn_global_load_async_to_lds_b128(gp1, lp1, 0, 0);
#else
    uint4 d0 = *(const uint4*)(g);
    uint4 d1 = *(const uint4*)(g + 8);
    *(uint4*)(l)     = d0;
    *(uint4*)(l + 8) = d1;
#endif
}

static __device__ __forceinline__ void wait_async_copies() {
#if HAVE_ASYNC_LDS
#if __has_builtin(__builtin_amdgcn_s_wait_asynccnt)
    __builtin_amdgcn_s_wait_asynccnt(0);
#else
    asm volatile("s_wait_asynccnt 0" ::: "memory");
#endif
#endif
}

static __device__ __forceinline__ unsigned short f2bf_bits(float f) {
    unsigned u = __builtin_bit_cast(unsigned, f);
    u += 0x7FFFu + ((u >> 16) & 1u);          // round-to-nearest-even
    return (unsigned short)(u >> 16);
}
static __device__ __forceinline__ __bf16 f2bf(float f) {
    unsigned short s = f2bf_bits(f);
    return __builtin_bit_cast(__bf16, s);
}
static __device__ __forceinline__ uint4 pack_bf16x8(floatx8 v) {
    uint4 r;
    r.x = (unsigned)f2bf_bits(v[0]) | ((unsigned)f2bf_bits(v[1]) << 16);
    r.y = (unsigned)f2bf_bits(v[2]) | ((unsigned)f2bf_bits(v[3]) << 16);
    r.z = (unsigned)f2bf_bits(v[4]) | ((unsigned)f2bf_bits(v[5]) << 16);
    r.w = (unsigned)f2bf_bits(v[6]) | ((unsigned)f2bf_bits(v[7]) << 16);
    return r;
}

struct Bf16FragBits { uint4 lo, hi; };
static __device__ __forceinline__ bf16x16 make_frag(uint4 lo, uint4 hi) {
    Bf16FragBits t{lo, hi};
    return __builtin_bit_cast(bf16x16, t);
}
static __device__ __forceinline__ floatx8 zero8() {
    floatx8 z = {0.f, 0.f, 0.f, 0.f, 0.f, 0.f, 0.f, 0.f};
    return z;
}

// A fragment (16x32 bf16, row-major LDS, element stride ld). ISA 7.12.2:
// lane r<16: row r, K = 0..7 & 16..23 ; lane r+16: row r, K = 8..15 & 24..31.
static __device__ __forceinline__ bf16x16 load_a(const __bf16* base, int ld, int lane) {
    const int r = lane & 15, h = lane >> 4;
    const __bf16* p = base + r * ld + 8 * h;
    uint4 lo = *(const uint4*)(p);
    uint4 hi = *(const uint4*)(p + 16);
    return make_frag(lo, hi);
}
// B fragment for NT GEMM (operand stored [N,K] row-major in LDS):
// lane: column n = lane&15, K = 8h..8h+7 & 16+8h..16+8h+7 (matches load_a).
static __device__ __forceinline__ bf16x16 load_b(const __bf16* base, int ld, int lane) {
    const int n = lane & 15, h = lane >> 4;
    const __bf16* p = base + n * ld + 8 * h;
    uint4 lo = *(const uint4*)(p);
    uint4 hi = *(const uint4*)(p + 16);
    return make_frag(lo, hi);
}

static __device__ __forceinline__ floatx8 wmma_bf16(bf16x16 a, bf16x16 b, floatx8 c) {
    return __builtin_amdgcn_wmma_f32_16x16x32_bf16(
        false, a, false, b, (short)0, c, false, false);
}

// ---------------------------------------------------------------------------
// fp32 -> bf16, 4 elements/thread
// ---------------------------------------------------------------------------
__global__ void __launch_bounds__(256)
cvt_f32_to_bf16(const float* __restrict__ src, __bf16* __restrict__ dst, int n) {
    int i = (blockIdx.x * blockDim.x + threadIdx.x) * 4;
    if (i < n) {
        float4 f = *(const float4*)(src + i);
        uint2 r;
        r.x = (unsigned)f2bf_bits(f.x) | ((unsigned)f2bf_bits(f.y) << 16);
        r.y = (unsigned)f2bf_bits(f.z) | ((unsigned)f2bf_bits(f.w) << 16);
        *(uint2*)(dst + i) = r;
    }
}

// ---------------------------------------------------------------------------
// NT GEMM: out[m,n] = sum_k A[m,k]*W[n,k] + bias[n]
// BM=128, BN=128, BK=32, 256 threads (8 waves, 4x2), wave tile 32x64.
// Double-buffered LDS with async global->LDS copies.
// MODE 0/1: bf16 [B,H,L,D] (Q/K); MODE 2: bf16 [B,H,D,L] (V^T);
// MODE 3: fp32 row-major [M,N] (final projection).
// ---------------------------------------------------------------------------
#define GEMM_BM 128
#define GEMM_BN 128
#define GEMM_BK 32
#define GEMM_LD 40   // padded LDS stride (80 B: 16B-aligned, conflict-free)

template <int MODE>
__global__ void __launch_bounds__(256)
gemm_nt_kernel(const __bf16* __restrict__ A, const __bf16* __restrict__ W,
               const float* __restrict__ bias, void* __restrict__ out,
               int M, int N, int K) {
    __shared__ __bf16 sA[2][GEMM_BM * GEMM_LD];
    __shared__ __bf16 sB[2][GEMM_BN * GEMM_LD];

    const int tid  = threadIdx.x;
    const int lane = tid & 31;
    const int wid  = tid >> 5;   // 0..7
    const int wm   = wid & 3;    // 4 waves across M (32 rows each)
    const int wn   = wid >> 2;   // 2 waves across N (64 cols each)
    const int m0   = blockIdx.y * GEMM_BM;
    const int n0   = blockIdx.x * GEMM_BN;

    floatx8 acc[2][4];
#pragma unroll
    for (int i = 0; i < 2; ++i)
#pragma unroll
        for (int j = 0; j < 4; ++j) acc[i][j] = zero8();

    // cooperative fill: 256 threads x 32B per tile
    const int lr = tid >> 1;
    const int lc = (tid & 1) * 16;
    const __bf16* gA = A + (size_t)(m0 + lr) * K + lc;
    const __bf16* gW = W + (size_t)(n0 + lr) * K + lc;
    const int lofs = lr * GEMM_LD + lc;

    const int nt = K / GEMM_BK;
    g2l_32B(&sA[0][lofs], gA);
    g2l_32B(&sB[0][lofs], gW);

    for (int kt = 0; kt < nt; ++kt) {
        wait_async_copies();
        __syncthreads();
        if (kt + 1 < nt) {
            g2l_32B(&sA[(kt + 1) & 1][lofs], gA + (kt + 1) * GEMM_BK);
            g2l_32B(&sB[(kt + 1) & 1][lofs], gW + (kt + 1) * GEMM_BK);
        }
        if (kt + 2 < nt) {
            __builtin_prefetch(gA + (kt + 2) * GEMM_BK, 0, 0);
            __builtin_prefetch(gW + (kt + 2) * GEMM_BK, 0, 0);
        }
        const __bf16* cA = sA[kt & 1];
        const __bf16* cB = sB[kt & 1];

        bf16x16 af[2];
#pragma unroll
        for (int i = 0; i < 2; ++i)
            af[i] = load_a(cA + (wm * 32 + i * 16) * GEMM_LD, GEMM_LD, lane);
        bf16x16 bf[4];
#pragma unroll
        for (int j = 0; j < 4; ++j)
            bf[j] = load_b(cB + (wn * 64 + j * 16) * GEMM_LD, GEMM_LD, lane);
#pragma unroll
        for (int i = 0; i < 2; ++i)
#pragma unroll
            for (int j = 0; j < 4; ++j)
                acc[i][j] = wmma_bf16(af[i], bf[j], acc[i][j]);
    }

    // Epilogue. C layout: elem e -> row m = e + 8*half, col n = lane&15.
    const int hh = lane >> 4;
    const int nl = lane & 15;
#pragma unroll
    for (int i = 0; i < 2; ++i) {
#pragma unroll
        for (int j = 0; j < 4; ++j) {
            const int mb = m0 + wm * 32 + i * 16 + 8 * hh;   // rows mb..mb+7
            const int n  = n0 + wn * 64 + j * 16 + nl;
            const float bv = bias[n];
            if (MODE == 2) {
                // V^T [B,H,D,L]: 8 consecutive rows -> 8 consecutive l; pack 16B
                const int b = mb >> 11, l = mb & 2047;
                const int h = n >> 6, d = n & 63;
                floatx8 t = acc[i][j] + bv;
                size_t idx = (((size_t)(b * 16 + h) * 64 + d) * 2048 + l);
                *(uint4*)((__bf16*)out + idx) = pack_bf16x8(t);
            } else if (MODE == 3) {
#pragma unroll
                for (int e = 0; e < 8; ++e)
                    ((float*)out)[(size_t)(mb + e) * N + n] = acc[i][j][e] + bv;
            } else {
                // Q/K [B,H,L,D]
#pragma unroll
                for (int e = 0; e < 8; ++e) {
                    const int m = mb + e;
                    const int b = m >> 11, l = m & 2047;
                    const int h = n >> 6, d = n & 63;
                    size_t idx = (((size_t)(b * 16 + h) * 2048 + l) * 64 + d);
                    ((__bf16*)out)[idx] = f2bf(acc[i][j][e] + bv);
                }
            }
        }
    }
}

// ---------------------------------------------------------------------------
// Flash attention: per (b,h); 64 Q-rows per block, 4 waves (16 rows/wave),
// 64-key blocks, online softmax, double-buffered K/V tiles.
// ---------------------------------------------------------------------------
#define FA_LD 72   // padded LDS stride (144 B: 16B-aligned, conflict-free)

__global__ void __launch_bounds__(128)
flash_attn_kernel(const __bf16* __restrict__ Q,   // [BH, L, D]
                  const __bf16* __restrict__ Km,  // [BH, L, D]
                  const __bf16* __restrict__ VT,  // [BH, D, L]
                  __bf16* __restrict__ Obf) {     // [B, L, E]
    __shared__ __bf16 sQ[64 * FA_LD];
    __shared__ __bf16 sK[2][64 * FA_LD];
    __shared__ __bf16 sV[2][64 * FA_LD];
    __shared__ __bf16 sP[4][16 * FA_LD];

    const int tid  = threadIdx.x;
    const int lane = tid & 31;
    const int w    = tid >> 5;          // wave 0..3
    const int bh   = blockIdx.y;        // 0..31
    const int q0   = blockIdx.x * 64;   // Q row block

    const size_t qkBase = (size_t)bh * 2048 * 64;
    const size_t vtBase = (size_t)bh * 64 * 2048;

    // cooperative fill pattern: thread -> row r, 32-elem half c
    const int fr = tid >> 1;
    const int fc = (tid & 1) * 32;
    const int fofs = fr * FA_LD + fc;

    {   // Q tile 64x64
        const __bf16* src = Q + qkBase + (size_t)(q0 + fr) * 64 + fc;
        g2l_32B(sQ + fofs, src);
        g2l_32B(sQ + fofs + 16, src + 16);
    }
    // issue K/V tile 0
    const __bf16* gK = Km + qkBase + (size_t)fr * 64 + fc;
    const __bf16* gV = VT + vtBase + (size_t)fr * 2048 + fc;
    g2l_32B(&sK[0][fofs], gK);
    g2l_32B(&sK[0][fofs + 16], gK + 16);
    g2l_32B(&sV[0][fofs], gV);
    g2l_32B(&sV[0][fofs + 16], gV + 16);

    wait_async_copies();
    __syncthreads();

    bf16x16 qf[2];
    qf[0] = load_a(sQ + (w * 16) * FA_LD + 0,  FA_LD, lane);
    qf[1] = load_a(sQ + (w * 16) * FA_LD + 32, FA_LD, lane);

    float row_max[8], row_sum[8];
#pragma unroll
    for (int i = 0; i < 8; ++i) { row_max[i] = -1e30f; row_sum[i] = 0.f; }
    floatx8 oacc[4];
#pragma unroll
    for (int j = 0; j < 4; ++j) oacc[j] = zero8();

    const float scale = 0.125f;  // 1/sqrt(64)
    const int hh = lane >> 4;
    const int nl = lane & 15;

    for (int kb = 0; kb < 32; ++kb) {
        wait_async_copies();
        __syncthreads();
        if (kb + 1 < 32) {   // issue next K/V block into other buffer
            const int nb = (kb + 1) & 1;
            const __bf16* nK = gK + (size_t)(kb + 1) * 64 * 64;
            const __bf16* nV = gV + (kb + 1) * 64;
            g2l_32B(&sK[nb][fofs], nK);
            g2l_32B(&sK[nb][fofs + 16], nK + 16);
            g2l_32B(&sV[nb][fofs], nV);
            g2l_32B(&sV[nb][fofs + 16], nV + 16);
        }
        const __bf16* cK = sK[kb & 1];
        const __bf16* cV = sV[kb & 1];

        // S = scale * (Q K^T): 16x64 per wave = 4 tiles
        floatx8 s[4];
#pragma unroll
        for (int ntl = 0; ntl < 4; ++ntl) {
            bf16x16 b0 = load_b(cK + (ntl * 16) * FA_LD + 0,  FA_LD, lane);
            bf16x16 b1 = load_b(cK + (ntl * 16) * FA_LD + 32, FA_LD, lane);
            floatx8 c = zero8();
            c = wmma_bf16(qf[0], b0, c);
            c = wmma_bf16(qf[1], b1, c);
            s[ntl] = c * scale;
        }

        // online softmax (row m = i + 8*half; the 16 lanes of a half share it)
        float factor[8];
#pragma unroll
        for (int i = 0; i < 8; ++i) {
            float mx = fmaxf(fmaxf(s[0][i], s[1][i]), fmaxf(s[2][i], s[3][i]));
            mx = fmaxf(mx, __shfl_xor(mx, 1));
            mx = fmaxf(mx, __shfl_xor(mx, 2));
            mx = fmaxf(mx, __shfl_xor(mx, 4));
            mx = fmaxf(mx, __shfl_xor(mx, 8));
            const float nm = fmaxf(row_max[i], mx);
            factor[i]  = __expf(row_max[i] - nm);
            row_max[i] = nm;
        }

        float psum[8];
#pragma unroll
        for (int i = 0; i < 8; ++i) psum[i] = 0.f;
#pragma unroll
        for (int ntl = 0; ntl < 4; ++ntl) {
#pragma unroll
            for (int i = 0; i < 8; ++i) {
                const float p = __expf(s[ntl][i] - row_max[i]);
                psum[i] += p;
                sP[w][(i + 8 * hh) * FA_LD + ntl * 16 + nl] = f2bf(p);
            }
        }
#pragma unroll
        for (int i = 0; i < 8; ++i) {
            float ps = psum[i];
            ps += __shfl_xor(ps, 1);
            ps += __shfl_xor(ps, 2);
            ps += __shfl_xor(ps, 4);
            ps += __shfl_xor(ps, 8);
            row_sum[i] = row_sum[i] * factor[i] + ps;
        }
#pragma unroll
        for (int j = 0; j < 4; ++j)
#pragma unroll
            for (int i = 0; i < 8; ++i) oacc[j][i] *= factor[i];

        // O += P V   (NT via V^T tiles; same-wave LDS ops are in-order)
        bf16x16 pf0 = load_a(sP[w] + 0,  FA_LD, lane);
        bf16x16 pf1 = load_a(sP[w] + 32, FA_LD, lane);
#pragma unroll
        for (int dt = 0; dt < 4; ++dt) {
            bf16x16 b0 = load_b(cV + (dt * 16) * FA_LD + 0,  FA_LD, lane);
            bf16x16 b1 = load_b(cV + (dt * 16) * FA_LD + 32, FA_LD, lane);
            oacc[dt] = wmma_bf16(pf0, b0, oacc[dt]);
            oacc[dt] = wmma_bf16(pf1, b1, oacc[dt]);
        }
    }

    // epilogue: normalize, write [B, L, E] bf16, e = h*64 + d
    const int b = bh >> 4, h = bh & 15;
#pragma unroll
    for (int dt = 0; dt < 4; ++dt) {
#pragma unroll
        for (int i = 0; i < 8; ++i) {
            const int lrow = q0 + w * 16 + i + 8 * hh;
            const int e    = h * 64 + dt * 16 + nl;
            const float v  = oacc[dt][i] / row_sum[i];
            Obf[(size_t)(b * 2048 + lrow) * 1024 + e] = f2bf(v);
        }
    }
}

// ---------------------------------------------------------------------------
// Launch
// ---------------------------------------------------------------------------
extern "C" void kernel_launch(void* const* d_in, const int* in_sizes, int n_in,
                              void* d_out, int out_size, void* d_ws, size_t ws_size,
                              hipStream_t stream) {
    (void)in_sizes; (void)n_in; (void)out_size; (void)ws_size;

    const float* x  = (const float*)d_in[0];
    const float* Wq = (const float*)d_in[1];
    const float* bq = (const float*)d_in[2];
    const float* Wk = (const float*)d_in[3];
    const float* bk = (const float*)d_in[4];
    const float* Wv = (const float*)d_in[5];
    const float* bv = (const float*)d_in[6];
    const float* Wo = (const float*)d_in[7];
    const float* bo = (const float*)d_in[8];

    char* ws = (char*)d_ws;
    __bf16* Xbf = (__bf16*)(ws + ((size_t)0  << 20));  // 8 MB  [4096,1024]
    __bf16* Wqb = (__bf16*)(ws + ((size_t)8  << 20));  // 2 MB
    __bf16* Wkb = (__bf16*)(ws + ((size_t)10 << 20));  // 2 MB
    __bf16* Wvb = (__bf16*)(ws + ((size_t)12 << 20));  // 2 MB
    __bf16* Wob = (__bf16*)(ws + ((size_t)14 << 20));  // 2 MB
    __bf16* Qb  = (__bf16*)(ws + ((size_t)16 << 20));  // 8 MB  [B,H,L,D]
    __bf16* Kb  = (__bf16*)(ws + ((size_t)24 << 20));  // 8 MB  [B,H,L,D]
    __bf16* VTb = (__bf16*)(ws + ((size_t)32 << 20));  // 8 MB  [B,H,D,L]
    __bf16* Ab  = (__bf16*)(ws + ((size_t)40 << 20));  // 8 MB  [B,L,E]

    const int NX = 4096 * 1024;
    const int NW = 1024 * 1024;

    cvt_f32_to_bf16<<<NX / 4 / 256, 256, 0, stream>>>(x,  Xbf, NX);
    cvt_f32_to_bf16<<<NW / 4 / 256, 256, 0, stream>>>(Wq, Wqb, NW);
    cvt_f32_to_bf16<<<NW / 4 / 256, 256, 0, stream>>>(Wk, Wkb, NW);
    cvt_f32_to_bf16<<<NW / 4 / 256, 256, 0, stream>>>(Wv, Wvb, NW);
    cvt_f32_to_bf16<<<NW / 4 / 256, 256, 0, stream>>>(Wo, Wob, NW);

    dim3 ggrid(1024 / GEMM_BN, 4096 / GEMM_BM);  // (8, 32)
    gemm_nt_kernel<0><<<ggrid, 256, 0, stream>>>(Xbf, Wqb, bq, Qb,  4096, 1024, 1024);
    gemm_nt_kernel<0><<<ggrid, 256, 0, stream>>>(Xbf, Wkb, bk, Kb,  4096, 1024, 1024);
    gemm_nt_kernel<2><<<ggrid, 256, 0, stream>>>(Xbf, Wvb, bv, VTb, 4096, 1024, 1024);

    flash_attn_kernel<<<dim3(2048 / 64, 32), 128, 0, stream>>>(Qb, Kb, VTb, Ab);

    gemm_nt_kernel<3><<<ggrid, 256, 0, stream>>>(Ab, Wob, bo, d_out, 4096, 1024, 1024);
}
